// QuantizedLinearInt4_48086453846080
// MI455X (gfx1250) — matrix operand order repaired and checked by reading, not compiled, not run on hardware
//
#include <hip/hip_runtime.h>

// QuantizedLinearInt4: out[8192,4096] = x[8192,4096] . dequant(W)[4096,4096]^T + bias
// bf16 WMMA (v_wmma_f32_16x16x32_bf16), f32 accumulation.
// Block tile 256x128 (8 waves, 4Mx2N), wave tile 64x64 (4x4 WMMA accums), K-chunk 64.
// Double-buffered LDS: stage chunk k+1 while computing chunk k; ONE barrier per chunk.
// Dequant: 256-entry LDS LUT (byte -> bf16 pair), valid since BN == GROUP_SIZE.

#define IN_F  4096
#define OUT_F 4096
#define BM 256
#define BN 128
#define BK 64
#define LDSS 72   // ushort stride per row: 64 data + 8 pad

typedef __attribute__((ext_vector_type(16))) __bf16 v16bf;
typedef __attribute__((ext_vector_type(8)))  __bf16 v8bf;
typedef __attribute__((ext_vector_type(2)))  __bf16 v2bf;
typedef __attribute__((ext_vector_type(8)))  float  v8f;

union Frag { v16bf v; v8bf h[2]; };

// Pack two f32 into one dword of bf16 (a -> low half, b -> high half).
__device__ __forceinline__ unsigned pk_bf16(float a, float b) {
#if __has_builtin(__builtin_amdgcn_cvt_pk_bf16_f32)
  v2bf p = __builtin_amdgcn_cvt_pk_bf16_f32(a, b);
  return __builtin_bit_cast(unsigned, p);
#else
  unsigned ua = __float_as_uint(a) + 0x8000u;  // round-to-nearest (ties away)
  unsigned ub = __float_as_uint(b) + 0x8000u;
  return __builtin_amdgcn_perm(ub, ua, 0x07060302u);
#endif
}

__global__ __launch_bounds__(256, 1) void qlinear_i4_wmma(
    const float* __restrict__ x,
    const int*   __restrict__ pw,      // packed bytes stored one-per-int32
    const float* __restrict__ scales,
    const float* __restrict__ zps,
    const float* __restrict__ bias,
    float* __restrict__ out)
{
  __shared__ unsigned short sX[2][BM * LDSS];  // 2 x 36 KB bf16 activations
  __shared__ unsigned short sW[2][BN * LDSS];  // 2 x 18 KB bf16 weights
  __shared__ unsigned       lutw[256];         // byte -> dequantized bf16 pair

  const int tid  = threadIdx.x;
  const int lane = tid & 31;
  const int wid  = tid >> 5;
  const int lm   = lane & 15;
  const int lh   = lane >> 4;
  const int wm   = (wid & 3) * 64;   // 4 waves over M
  const int wn   = (wid >> 2) * 64;  // 2 waves over N

  const int bm = blockIdx.x * BM;
  const int bn = blockIdx.y * BN;
  const int g  = blockIdx.y;         // BN == GROUP_SIZE -> one quant group per block

  const float s  = scales[g];
  const float c0 = -zps[g] * s;      // w = q*s + c0

  // Build dequant LUT: one entry per thread (blockDim == 256).
  {
    int p   = tid;
    int qlo = (p << 28) >> 28;       // low nibble = even K
    int qhi = (p << 24) >> 28;       // high nibble = odd K
    lutw[tid] = pk_bf16(fmaf((float)qlo, s, c0), fmaf((float)qhi, s, c0));
  }

  // Loader mapping: x -> one full K-chunk row per thread (256 rows).
  //                 w -> (row, half) per thread (128 rows x 2 halves).
  const int wlr   = tid >> 1;        // 0..127
  const int whalf = tid & 1;         // 0..1

  const float* xg = x  + (size_t)(bm + tid) * IN_F;
  const int*   wg = pw + (size_t)(bn + wlr) * (IN_F / 2) + whalf * 16;

  float4 xr[16];  // 64 f32 activations (one chunk-row)
  int4   wr[4];   // 16 packed bytes = 32 int4 weights

  auto load_chunk = [&](int kc) {
    const float4* xp = (const float4*)(xg + kc * BK);
#pragma unroll
    for (int i = 0; i < 16; ++i) xr[i] = xp[i];
    const int4* wp = (const int4*)(wg + kc * (BK / 2));
#pragma unroll
    for (int i = 0; i < 4; ++i) wr[i] = wp[i];
  };

  auto stage_to_lds = [&](int buf) {
    // activations: f32 -> packed bf16, full 64-element row
    union { unsigned u[32]; uint4 q[8]; } tx;
    const float* xv = (const float*)xr;
#pragma unroll
    for (int i = 0; i < 32; ++i) tx.u[i] = pk_bf16(xv[2 * i], xv[2 * i + 1]);
    uint4* dx = (uint4*)&sX[buf][tid * LDSS];
#pragma unroll
    for (int i = 0; i < 8; ++i) dx[i] = tx.q[i];

    // weights: LUT lookup (byte -> bf16 pair)
    union { unsigned u[16]; uint4 q[4]; } tw;
    const int* wv = (const int*)wr;
#pragma unroll
    for (int i = 0; i < 16; ++i) tw.u[i] = lutw[wv[i] & 0xFF];
    uint4* dw = (uint4*)&sW[buf][wlr * LDSS + whalf * 32];
#pragma unroll
    for (int i = 0; i < 4; ++i) dw[i] = tw.q[i];
  };

  v8f acc[4][4];
#pragma unroll
  for (int mt = 0; mt < 4; ++mt)
#pragma unroll
    for (int nt = 0; nt < 4; ++nt)
      acc[mt][nt] = (v8f){0.f, 0.f, 0.f, 0.f, 0.f, 0.f, 0.f, 0.f};

  auto compute = [&](int buf) {
#pragma unroll
    for (int ks = 0; ks < BK; ks += 32) {
      Frag a[4], b[4];
      // A frag (16x32 bf16): lane lm = row; chunks [lh*8,+8) and [16+lh*8,+8)
#pragma unroll
      for (int mt = 0; mt < 4; ++mt) {
        const unsigned short* base = &sX[buf][(wm + mt * 16 + lm) * LDSS + ks + lh * 8];
        a[mt].h[0] = *(const v8bf*)(base);
        a[mt].h[1] = *(const v8bf*)(base + 16);
      }
      // B frag (32x16 bf16): lane lm = col; chunks [lh*8,+8) and [16+lh*8,+8)
#pragma unroll
      for (int nt = 0; nt < 4; ++nt) {
        const unsigned short* base = &sW[buf][(wn + nt * 16 + lm) * LDSS + ks + lh * 8];
        b[nt].h[0] = *(const v8bf*)(base);
        b[nt].h[1] = *(const v8bf*)(base + 16);
      }
#pragma unroll
      for (int mt = 0; mt < 4; ++mt)
#pragma unroll
        for (int nt = 0; nt < 4; ++nt)
          acc[mt][nt] = __builtin_amdgcn_wmma_f32_16x16x32_bf16(
              false, a[mt].v, false, b[nt].v, (short)0, acc[mt][nt], false, false);
    }
  };

  const int KCH = IN_F / BK;   // 64 chunks

  load_chunk(0);
  __syncthreads();             // LUT ready
  stage_to_lds(0);
  __syncthreads();             // buffer 0 ready

  for (int kc = 0; kc < KCH; ++kc) {
    const int cur = kc & 1;
    if (kc + 1 < KCH) load_chunk(kc + 1);     // HBM fetch overlaps WMMA
    compute(cur);                             // WMMA on current buffer
    if (kc + 1 < KCH) stage_to_lds(cur ^ 1);  // stage next buffer (no reader conflict)
    __syncthreads();                          // single barrier per chunk
  }

  // Epilogue: lane (lm) = N, VGPR v -> M = lh*8 + v. Add bias, store f32.
#pragma unroll
  for (int nt = 0; nt < 4; ++nt) {
    const int gn = bn + wn + nt * 16 + lm;
    const float bv = bias[gn];
#pragma unroll
    for (int mt = 0; mt < 4; ++mt) {
      const int gm0 = bm + wm + mt * 16 + lh * 8;
#pragma unroll
      for (int v = 0; v < 8; ++v)
        out[(size_t)(gm0 + v) * OUT_F + gn] = acc[mt][nt][v] + bv;
    }
  }
}

extern "C" void kernel_launch(void* const* d_in, const int* in_sizes, int n_in,
                              void* d_out, int out_size, void* d_ws, size_t ws_size,
                              hipStream_t stream) {
  const float* x    = (const float*)d_in[0];
  const int*   pw   = (const int*)d_in[1];
  const float* sc   = (const float*)d_in[2];
  const float* zp   = (const float*)d_in[3];
  const float* bias = (const float*)d_in[4];
  float* out = (float*)d_out;

  const int M = in_sizes[0] / IN_F;          // 8192
  dim3 grid(M / BM, OUT_F / BN);             // 32 x 32
  qlinear_i4_wmma<<<grid, 256, 0, stream>>>(x, pw, sc, zp, bias, out);
}
